// GptOssAttention_13408887898142
// MI455X (gfx1250) — compile-verified, matrix operand order checked by reading, not run on hardware
//
#include <hip/hip_runtime.h>
#include <hip/hip_bf16.h>

// ---------------------------------------------------------------------------
// Types for CDNA5 WMMA (wave32): v_wmma_f32_16x16x32_bf16
// ---------------------------------------------------------------------------
typedef __attribute__((ext_vector_type(16))) __bf16 v16bf;
typedef __attribute__((ext_vector_type(8)))  __bf16 v8bf;
typedef __attribute__((ext_vector_type(8)))  float  v8f;

__device__ __forceinline__ unsigned short f2bf(float f) {
  unsigned u = __float_as_uint(f);
  u += 0x7FFFu + ((u >> 16) & 1u);           // round-to-nearest-even
  return (unsigned short)(u >> 16);
}
__device__ __forceinline__ float bf2f(unsigned short h) {
  return __uint_as_float(((unsigned)h) << 16);
}

#define HID        2880
#define HD         64
#define NQH        64
#define NKVH       8
#define GRP        8
#define WIN        128
#define QSZ        4096
#define KVSZ       512
#define QKVD       5120

// LDS row strides in halves (row base stays 16B aligned)
#define LROW 48    // GEMM tiles: 32 data + 16 pad
#define KROW 72    // attention K tile: 64 data + 8 pad (144B = 9*16)
#define VROW 264   // attention V tile: 256 data + 8 pad (528B = 33*16)
#define PROW 136   // attention P tile: 128 data + 8 pad (272B = 17*16)

// ---------------------------------------------------------------------------
// Generic C = A @ W^T + bias GEMM.
//   A: [M][K]  (f32 or bf16, selected by ABF)
//   W: [N][K]  f32 (row-major weights; row n is output column n)
//   C: [M][N]  (f32 or bf16, selected by CBF)
// Workgroup: 256 threads = 8 waves; tile 128(M) x 64(N); wave tile 32x32.
// ---------------------------------------------------------------------------
template<bool ABF, bool CBF>
__global__ __launch_bounds__(256)
void gemm_ws(const void* __restrict__ Ap, const float* __restrict__ W,
             const float* __restrict__ bias, void* __restrict__ Cp,
             int M, int N, int K)
{
  __shared__ alignas(16) unsigned short lA[128 * LROW];
  __shared__ alignas(16) unsigned short lW[64 * LROW];

  const int tid  = threadIdx.x;
  const int lane = tid & 31;
  const int wave = tid >> 5;
  const int wm   = wave & 3;         // 4 waves along M
  const int wn   = wave >> 2;        // 2 waves along N
  const int n0   = blockIdx.x * 64;
  const int m0   = blockIdx.y * 128;
  const int l16  = lane & 15;
  const int hf   = lane >> 4;

  v8f acc[2][2];
#pragma unroll
  for (int i = 0; i < 2; i++)
#pragma unroll
    for (int j = 0; j < 2; j++) {
      float b = bias[n0 + wn * 32 + j * 16 + l16];
#pragma unroll
      for (int r = 0; r < 8; r++) acc[i][j][r] = b;
    }

  const float*          Af = (const float*)Ap;
  const unsigned short* Ab = (const unsigned short*)Ap;

  for (int k0 = 0; k0 < K; k0 += 32) {
    // ---- stage A tile: 128 x 32 -> bf16 LDS -------------------------------
    if constexpr (ABF) {
#pragma unroll
      for (int c = 0; c < 2; c++) {
        int idx = tid + c * 256;                 // 512 chunks of 8 halves
        int row = idx >> 2;
        int col = (idx & 3) * 8;
        v8bf v = *(const v8bf*)(Ab + (size_t)(m0 + row) * K + k0 + col);
        *(v8bf*)(lA + row * LROW + col) = v;
      }
    } else {
#pragma unroll
      for (int c = 0; c < 4; c++) {
        int idx = tid + c * 256;                 // 1024 float4 chunks
        int row = idx >> 3;
        int col = (idx & 7) * 4;
        float4 f4 = *(const float4*)(Af + (size_t)(m0 + row) * K + k0 + col);
        unsigned v01 = (unsigned)f2bf(f4.x) | ((unsigned)f2bf(f4.y) << 16);
        unsigned v23 = (unsigned)f2bf(f4.z) | ((unsigned)f2bf(f4.w) << 16);
        unsigned* d = (unsigned*)(lA + row * LROW + col);
        d[0] = v01; d[1] = v23;
      }
    }
    // ---- stage W tile: 64 x 32 f32 -> bf16 LDS ----------------------------
#pragma unroll
    for (int c = 0; c < 2; c++) {
      int idx = tid + c * 256;                   // 512 float4 chunks
      int row = idx >> 3;
      int col = (idx & 7) * 4;
      float4 f4 = *(const float4*)(W + (size_t)(n0 + row) * K + k0 + col);
      unsigned v01 = (unsigned)f2bf(f4.x) | ((unsigned)f2bf(f4.y) << 16);
      unsigned v23 = (unsigned)f2bf(f4.z) | ((unsigned)f2bf(f4.w) << 16);
      unsigned* d = (unsigned*)(lW + row * LROW + col);
      d[0] = v01; d[1] = v23;
    }
    // ---- prefetch next K tile into cache (global_prefetch_b8) -------------
    if (k0 + 32 < K) {
      const char* abase = (const char*)Ap;
      const size_t esz = ABF ? 2 : 4;
      int row = tid & 127;
      __builtin_prefetch(abase + ((size_t)(m0 + row) * K + k0 + 32) * esz, 0, 0);
      if (tid < 64)
        __builtin_prefetch((const char*)(W + (size_t)(n0 + tid) * K + k0 + 32), 0, 0);
    }
    __syncthreads();

    // ---- fragments (two 16B ds_load_b128 each, per CDNA5 wave32 layout) ---
    v16bf afr[2], wfr[2];
#pragma unroll
    for (int i = 0; i < 2; i++) {
      const unsigned short* rp = lA + (wm * 32 + i * 16 + l16) * LROW;
      v8bf lo = *(const v8bf*)(rp + hf * 8);
      v8bf hi = *(const v8bf*)(rp + 16 + hf * 8);
#pragma unroll
      for (int e = 0; e < 8; e++) { afr[i][e] = lo[e]; afr[i][8 + e] = hi[e]; }
    }
#pragma unroll
    for (int j = 0; j < 2; j++) {
      const unsigned short* rp = lW + (wn * 32 + j * 16 + l16) * LROW;
      v8bf lo = *(const v8bf*)(rp + hf * 8);
      v8bf hi = *(const v8bf*)(rp + 16 + hf * 8);
#pragma unroll
      for (int e = 0; e < 8; e++) { wfr[j][e] = lo[e]; wfr[j][8 + e] = hi[e]; }
    }
#pragma unroll
    for (int i = 0; i < 2; i++)
#pragma unroll
      for (int j = 0; j < 2; j++)
        acc[i][j] = __builtin_amdgcn_wmma_f32_16x16x32_bf16(
            false, afr[i], false, wfr[j], (short)0, acc[i][j], false, false);
    __syncthreads();
  }

  // ---- store: C VGPR r holds M = hf*8 + r, N = l16 -------------------------
#pragma unroll
  for (int i = 0; i < 2; i++)
#pragma unroll
    for (int j = 0; j < 2; j++) {
      int col = n0 + wn * 32 + j * 16 + l16;
#pragma unroll
      for (int r = 0; r < 8; r++) {
        int row = m0 + wm * 32 + i * 16 + hf * 8 + r;
        if constexpr (CBF)
          ((unsigned short*)Cp)[(size_t)row * N + col] = f2bf(acc[i][j][r]);
        else
          ((float*)Cp)[(size_t)row * N + col] = acc[i][j][r];
      }
    }
}

// ---------------------------------------------------------------------------
// YaRN RoPE + repack: qkv(bf16 [n][5120]) ->
//   qb [n][64*64] bf16 (rope, * sm_scale)
//   kb [n][8*64]  bf16 (rope)
//   vt [8][64][n] bf16 (transposed V for PV b-fragments)
// One block per token.
// ---------------------------------------------------------------------------
__global__ __launch_bounds__(256)
void rope_pack(const unsigned short* __restrict__ qkv,
               const int* __restrict__ pos_ids,
               unsigned short* __restrict__ qb,
               unsigned short* __restrict__ kb,
               unsigned short* __restrict__ vt,
               int n)
{
  const int tok = blockIdx.x;
  const int tid = threadIdx.x;
  const float pos = (float)pos_ids[tok];
  const unsigned short* row = qkv + (size_t)tok * QKVD;

  const float two_pi  = 6.2831853071795865f;
  const float logbase = __logf(150000.0f);
  const float low  = 32.0f * __logf(4096.0f / (32.0f * two_pi)) / logbase;
  const float high = 32.0f * __logf(4096.0f / two_pi) / logbase;
  const float conc = 0.1f * __logf(32.0f) + 1.0f;

  for (int item = tid; item < 72 * 32; item += 256) {
    int h = item >> 5;
    int f = item & 31;
    float freq   = __powf(150000.0f, (float)(2 * f) * (1.0f / 64.0f));
    float interp = 1.0f / (32.0f * freq);
    float extrap = 1.0f / freq;
    float ramp   = ((float)f - low) / (high - low);
    float mk     = 1.0f - fminf(fmaxf(ramp, 0.0f), 1.0f);
    float inv    = interp * (1.0f - mk) + extrap * mk;
    float ang    = pos * inv;
    float c = __cosf(ang) * conc;
    float s = __sinf(ang) * conc;
    if (h < NQH) {
      float x1 = bf2f(row[h * HD + f]);
      float x2 = bf2f(row[h * HD + f + 32]);
      qb[(size_t)tok * QSZ + h * HD + f]      = f2bf((x1 * c - x2 * s) * 0.125f);
      qb[(size_t)tok * QSZ + h * HD + f + 32] = f2bf((x2 * c + x1 * s) * 0.125f);
    } else {
      int hk = h - NQH;
      float x1 = bf2f(row[QSZ + hk * HD + f]);
      float x2 = bf2f(row[QSZ + hk * HD + f + 32]);
      kb[(size_t)tok * KVSZ + hk * HD + f]      = f2bf(x1 * c - x2 * s);
      kb[(size_t)tok * KVSZ + hk * HD + f + 32] = f2bf(x2 * c + x1 * s);
    }
  }
  for (int e = tid; e < KVSZ; e += 256) {
    int g = e >> 6;
    int d = e & 63;
    vt[((size_t)(g * HD + d)) * n + tok] = row[QSZ + KVSZ + e];
  }
}

// ---------------------------------------------------------------------------
// Windowed attention, flash-style online softmax.
// grid = nb*8 (block, kv-head); 8 waves = 8 q-heads; 16 query rows per step.
// K (256x64) and V^T (64x256) tiles are staged ONCE per workgroup into LDS
// via CDNA5 async-to-LDS copies (ASYNCcnt path, no VGPR round-trip), shared
// by all 8 q-head waves. An opaque zero offset inside the qc loop stops LICM
// from hoisting the qc-invariant LDS fragment loads.
// ---------------------------------------------------------------------------
__global__ __launch_bounds__(256)
void attn_win(const unsigned short* __restrict__ qb,
              const unsigned short* __restrict__ kb,
              const unsigned short* __restrict__ vt,
              const float* __restrict__ sinks,
              unsigned short* __restrict__ attn,   // bf16 [n][4096]
              int n)
{
  __shared__ alignas(16) unsigned short ksh[256 * KROW];   // [key][dim]
  __shared__ alignas(16) unsigned short vsh[64 * VROW];    // [dim][key]
  __shared__ alignas(16) unsigned short pls[8][16 * PROW]; // per-wave P half

  const int tid  = threadIdx.x;
  const int blk  = blockIdx.x >> 3;
  const int g    = blockIdx.x & 7;
  const int wave = tid >> 5;
  const int lane = tid & 31;
  const int l16  = lane & 15;
  const int hf   = lane >> 4;
  const int qh   = g * GRP + wave;
  const float sink = sinks[qh];
  const int keybase = blk * WIN - WIN;          // token of key col 0 (maybe <0)
  unsigned short* pbuf = &pls[wave][0];

  // ---- async stage: K tile 256x64 (row-major) -----------------------------
#pragma unroll
  for (int c = 0; c < 8; c++) {
    int idx = tid + c * 256;                    // 2048 chunks of 8 halves
    int kk  = idx >> 3;                         // key 0..255
    int col = (idx & 7) * 8;
    int kt  = keybase + kk;
    if (kt < 0) kt = 0;                         // masked later
    const unsigned short* src = kb + ((size_t)kt * NKVH + g) * HD + col;
    unsigned ldsa = (unsigned)(size_t)(ksh + kk * KROW + col);
    asm volatile("global_load_async_to_lds_b128 %0, %1, off"
                 :: "v"(ldsa), "v"(src) : "memory");
  }
  // ---- async stage: V^T tile 64x256 (dim-major) ---------------------------
#pragma unroll
  for (int c = 0; c < 8; c++) {
    int idx = tid + c * 256;                    // 2048 chunks of 8 halves
    int d   = idx >> 5;                         // dim 0..63
    int col = (idx & 31) * 8;                   // key 0..255 (chunk start)
    int kt  = keybase + col;
    if (kt < 0) kt = 0;                         // masked later
    const unsigned short* src = vt + ((size_t)(g * HD + d)) * n + kt;
    unsigned ldsa = (unsigned)(size_t)(vsh + d * VROW + col);
    asm volatile("global_load_async_to_lds_b128 %0, %1, off"
                 :: "v"(ldsa), "v"(src) : "memory");
  }
  asm volatile("s_wait_asynccnt 0" ::: "memory");
  __syncthreads();

  for (int qc = 0; qc < 8; qc++) {
    // Opaque zero: prevents LICM of the qc-invariant LDS fragment loads.
    int zoff = 0;
    asm volatile("" : "+v"(zoff));

    // ---- A fragments of Q: row = token, K = head_dim (2 x 32) --------------
    const unsigned short* qrow =
        qb + ((size_t)(blk * WIN + qc * 16 + l16) * NQH + qh) * HD;
    v16bf aq[2];
#pragma unroll
    for (int kk = 0; kk < 2; kk++) {
      v8bf lo = *(const v8bf*)(qrow + kk * 32 + hf * 8);
      v8bf hi = *(const v8bf*)(qrow + kk * 32 + 16 + hf * 8);
#pragma unroll
      for (int e = 0; e < 8; e++) { aq[kk][e] = lo[e]; aq[kk][8 + e] = hi[e]; }
    }

    // running softmax state per row r (rows i = qc*16 + hf*8 + r)
    float mrun[8], srun[8];
#pragma unroll
    for (int r = 0; r < 8; r++) { mrun[r] = sink; srun[r] = 0.0f; }
    v8f oacc[4];
#pragma unroll
    for (int t = 0; t < 4; t++) { v8f z = {}; oacc[t] = z; }

#pragma unroll
    for (int h2 = 0; h2 < 2; h2++) {
      // ---- scores for this half: 8 col-tiles of 16 keys -------------------
      v8f e8[8];
#pragma unroll
      for (int nt = 0; nt < 8; nt++) {
        const unsigned short* krow =
            ksh + (h2 * 128 + nt * 16 + l16) * KROW + zoff;
        v16bf bk0, bk1;
        {
          v8bf lo = *(const v8bf*)(krow + hf * 8);
          v8bf hi = *(const v8bf*)(krow + 16 + hf * 8);
#pragma unroll
          for (int e = 0; e < 8; e++) { bk0[e] = lo[e]; bk0[8 + e] = hi[e]; }
        }
        {
          v8bf lo = *(const v8bf*)(krow + 32 + hf * 8);
          v8bf hi = *(const v8bf*)(krow + 48 + hf * 8);
#pragma unroll
          for (int e = 0; e < 8; e++) { bk1[e] = lo[e]; bk1[8 + e] = hi[e]; }
        }
        v8f z = {};
        z = __builtin_amdgcn_wmma_f32_16x16x32_bf16(false, aq[0], false, bk0,
                                                    (short)0, z, false, false);
        e8[nt] = __builtin_amdgcn_wmma_f32_16x16x32_bf16(false, aq[1], false, bk1,
                                                         (short)0, z, false, false);
        __builtin_amdgcn_sched_barrier(0);      // limit live B fragments
      }

      // ---- mask + tile row max -------------------------------------------
      float mh[8];
#pragma unroll
      for (int r = 0; r < 8; r++) mh[r] = -1e30f;
#pragma unroll
      for (int nt = 0; nt < 8; nt++) {
        int jj = h2 * 128 + nt * 16 + l16;
#pragma unroll
        for (int r = 0; r < 8; r++) {
          int i = qc * 16 + hf * 8 + r;
          bool ok = (jj >= i + 1) && (jj <= i + WIN) && ((blk > 0) || (jj >= WIN));
          float v = ok ? e8[nt][r] : -1e30f;
          e8[nt][r] = v;
          mh[r] = fmaxf(mh[r], v);
        }
      }
#pragma unroll
      for (int r = 0; r < 8; r++) {
#pragma unroll
        for (int m = 1; m < 16; m <<= 1)
          mh[r] = fmaxf(mh[r], __shfl_xor(mh[r], m, 32));
      }

      // ---- online rescale of running state --------------------------------
      float scl[8];
#pragma unroll
      for (int r = 0; r < 8; r++) {
        float mnew = fmaxf(mrun[r], mh[r]);
        scl[r] = __expf(mrun[r] - mnew);
        mrun[r] = mnew;
        srun[r] *= scl[r];
      }
#pragma unroll
      for (int t = 0; t < 4; t++)
#pragma unroll
        for (int r = 0; r < 8; r++) oacc[t][r] *= scl[r];

      // ---- exp, row-sum, write P half to LDS ------------------------------
#pragma unroll
      for (int nt = 0; nt < 8; nt++)
#pragma unroll
        for (int r = 0; r < 8; r++) {
          float p = __expf(e8[nt][r] - mrun[r]);
          srun[r] += p;
          pbuf[(hf * 8 + r) * PROW + nt * 16 + l16] = f2bf(p);
        }

      // ---- PV for this half: O += P(16x128) @ V(128x64) -------------------
#pragma unroll
      for (int ks = 0; ks < 4; ks++) {
        v16bf ap;
        {
          const unsigned short* rp = pbuf + l16 * PROW + ks * 32;
          v8bf lo = *(const v8bf*)(rp + hf * 8);
          v8bf hi = *(const v8bf*)(rp + 16 + hf * 8);
#pragma unroll
          for (int e = 0; e < 8; e++) { ap[e] = lo[e]; ap[8 + e] = hi[e]; }
        }
#pragma unroll
        for (int t = 0; t < 4; t++) {
          int d = t * 16 + l16;
          const unsigned short* vrow = vsh + d * VROW + zoff;
          int c0 = h2 * 128 + ks * 32 + hf * 8;
          v8bf lo = *(const v8bf*)(vrow + c0);
          v8bf hi = *(const v8bf*)(vrow + c0 + 16);
          v16bf bv;
#pragma unroll
          for (int e = 0; e < 8; e++) { bv[e] = lo[e]; bv[8 + e] = hi[e]; }
          oacc[t] = __builtin_amdgcn_wmma_f32_16x16x32_bf16(
              false, ap, false, bv, (short)0, oacc[t], false, false);
        }
        __builtin_amdgcn_sched_barrier(0);      // limit live B fragments
      }
    }

    // ---- finalize: divide by denom (incl. sink term) and store ------------
    float inv[8];
#pragma unroll
    for (int r = 0; r < 8; r++)
      inv[r] = 1.0f / (srun[r] + __expf(sink - mrun[r]));
#pragma unroll
    for (int t = 0; t < 4; t++) {
      int col = qh * HD + t * 16 + l16;
#pragma unroll
      for (int r = 0; r < 8; r++) {
        int row = blk * WIN + qc * 16 + hf * 8 + r;
        attn[(size_t)row * QSZ + col] = f2bf(oacc[t][r] * inv[r]);
      }
    }
  }
}

// ---------------------------------------------------------------------------
// Launcher
// ---------------------------------------------------------------------------
extern "C" void kernel_launch(void* const* d_in, const int* in_sizes, int n_in,
                              void* d_out, int out_size, void* d_ws, size_t ws_size,
                              hipStream_t stream)
{
  const float* hidden = (const float*)d_in[0];
  const int*   pos    = (const int*)d_in[1];
  const float* qkv_w  = (const float*)d_in[2];
  const float* qkv_b  = (const float*)d_in[3];
  const float* o_w    = (const float*)d_in[4];
  const float* o_b    = (const float*)d_in[5];
  const float* sinks  = (const float*)d_in[6];
  float*       out    = (float*)d_out;

  const int n  = in_sizes[0] / HID;   // 8192
  const int nb = n / WIN;             // 64

  char* ws = (char*)d_ws;
  size_t off = 0;
  unsigned short* qkv = (unsigned short*)(ws + off); off += (size_t)n * QKVD * 2;
  unsigned short* qb  = (unsigned short*)(ws + off); off += (size_t)n * QSZ  * 2;
  unsigned short* kb  = (unsigned short*)(ws + off); off += (size_t)n * KVSZ * 2;
  unsigned short* vt  = (unsigned short*)(ws + off); off += (size_t)n * KVSZ * 2;
  unsigned short* at  = (unsigned short*)(ws + off); off += (size_t)n * QSZ  * 2;

  // 1) qkv = hidden @ qkv_w^T + qkv_b   (f32 in, bf16 out)
  dim3 g1(QKVD / 64, n / 128);
  gemm_ws<false, true><<<g1, 256, 0, stream>>>(hidden, qkv_w, qkv_b, qkv,
                                               n, QKVD, HID);
  // 2) rope + repack
  rope_pack<<<n, 256, 0, stream>>>(qkv, pos, qb, kb, vt, n);
  // 3) windowed attention
  attn_win<<<nb * NKVH, 256, 0, stream>>>(qb, kb, vt, sinks, at, n);
  // 4) out = attn @ o_w^T + o_b   (bf16 in, f32 out)
  dim3 g2(HID / 64, n / 128);
  gemm_ws<true, false><<<g2, 256, 0, stream>>>(at, o_w, o_b, out,
                                               n, HID, QSZ);
}
